// VectorQuantizer_79130477462064
// MI455X (gfx1250) — matrix operand order repaired and checked by reading, not compiled, not run on hardware
//
#include <hip/hip_runtime.h>
#include <hip/hip_bf16.h>
#include <math.h>

// ---------------- problem constants ----------------
#define NPTS   32768          // 8 * 16*16*16 points
#define KCODES 1024
#define CDIM   256
#define SPAT   4096           // 16*16*16 spatial per batch
#define TOTELM 8388608        // NPTS * CDIM

// ---------------- tiling ----------------
#define PTS_PER_BLOCK 128     // 4 waves * 2 M-tiles * 16 rows
#define NCHUNK        64      // codes staged in LDS per iteration (4 N-tiles)
#define ZSTRIDE       264     // padded bf16 stride (multiple of 8 -> 16B aligned rows)

typedef __attribute__((ext_vector_type(16))) __bf16 v16bf;
typedef __attribute__((ext_vector_type(8)))  __bf16 v8bf;
typedef __attribute__((ext_vector_type(8)))  float  v8f;

// ---------------- workspace layout (bytes) ----------------
#define WS_EMB_BF   0          // 1024*256*2 = 524288
#define WS_NORMS_E  524288     // 1024 f32
#define WS_SE       528384     // 256 f32 (direct write)
#define WS_SE_TOT   529408     // 1 f32  (direct write)
#define WS_ZERO0    532480     // ---- zeroed region start ----
#define WS_COUNTS   532480     // 1024 u32
#define WS_SZ       536576     // 256 f32 (atomic)
#define WS_SZ_TOT   537600     // 1 f32  (atomic)
#define WS_LOSS     537604     // 1 f32  (atomic)
#define WS_ZERO_LEN (537608 - 532480)

// ============================================================
// Kernel 1: per-code ||e||^2 and bf16 copy of codebook
// ============================================================
__global__ __launch_bounds__(256) void prep_emb(const float* __restrict__ emb,
                                                __bf16* __restrict__ emb_bf,
                                                float* __restrict__ norms_e) {
    __shared__ float red[256];
    const int j = blockIdx.x;          // code
    const int c = threadIdx.x;         // channel
    float v = emb[(size_t)j * CDIM + c];
    emb_bf[(size_t)j * CDIM + c] = (__bf16)v;
    red[c] = v * v;
    __syncthreads();
    for (int s = 128; s > 0; s >>= 1) {
        if (c < s) red[c] += red[c + s];
        __syncthreads();
    }
    if (c == 0) norms_e[j] = red[0];
}

// ============================================================
// Kernel 2: column sums of emb (se[c]) and total sum of squares (Se)
// ============================================================
__global__ __launch_bounds__(256) void emb_stats(const float* __restrict__ emb,
                                                 float* __restrict__ se,
                                                 float* __restrict__ Se) {
    __shared__ float red[256];
    const int c = threadIdx.x;
    float s = 0.f, sq = 0.f;
    for (int j = 0; j < KCODES; ++j) {
        float v = emb[(size_t)j * CDIM + c];
        s += v; sq += v * v;
    }
    se[c] = s;
    red[c] = sq;
    __syncthreads();
    for (int st = 128; st > 0; st >>= 1) {
        if (c < st) red[c] += red[c + st];
        __syncthreads();
    }
    if (c == 0) Se[0] = red[0];
}

// ============================================================
// Kernel 3: per-channel sums of z (sz[c]) and total sum of squares (Sz)
// grid = 8*256 (b,c) pairs
// ============================================================
__global__ __launch_bounds__(256) void z_stats(const float* __restrict__ z,
                                               float* __restrict__ sz,
                                               float* __restrict__ Sz) {
    __shared__ float rs[256];
    __shared__ float rq[256];
    const int bc = blockIdx.x;
    const int c = bc & 255;
    const float* row = z + (size_t)bc * SPAT;
    float s = 0.f, sq = 0.f;
    for (int i = threadIdx.x; i < SPAT; i += 256) {
        float v = row[i];
        s += v; sq += v * v;
    }
    rs[threadIdx.x] = s; rq[threadIdx.x] = sq;
    __syncthreads();
    for (int st = 128; st > 0; st >>= 1) {
        if (threadIdx.x < st) { rs[threadIdx.x] += rs[threadIdx.x + st];
                                rq[threadIdx.x] += rq[threadIdx.x + st]; }
        __syncthreads();
    }
    if (threadIdx.x == 0) { atomicAdd(&sz[c], rs[0]); atomicAdd(Sz, rq[0]); }
}

// ============================================================
// Kernel 4: main VQ — WMMA distance argmin + gather + loss + histogram
// block = 128 threads (4 waves); each wave owns 2 M-tiles (32 points);
// NCHUNK=64 codes staged per iteration (4 N-tiles per wave).
// grid = NPTS/128 = 256
// ============================================================
__global__ __launch_bounds__(128) void vq_main(const float* __restrict__ z,
                                               const float* __restrict__ emb,
                                               const __bf16* __restrict__ emb_bf,
                                               const float* __restrict__ norms_e,
                                               float* __restrict__ zq_out,
                                               float* __restrict__ idx_out,
                                               unsigned int* __restrict__ counts,
                                               float* __restrict__ loss_acc) {
    __shared__ __bf16 z_lds[PTS_PER_BLOCK * ZSTRIDE];   // 67584 B
    __shared__ __bf16 e_lds[NCHUNK * ZSTRIDE];          // 33792 B
    __shared__ int    idx_lds[PTS_PER_BLOCK];
    __shared__ float  red_lds[128];

    const int tid = threadIdx.x;
    const int g0  = blockIdx.x * PTS_PER_BLOCK;   // first global point
    const int b   = g0 >> 12;                     // batch
    const int s0  = g0 & (SPAT - 1);              // spatial offset (block stays in one batch)
    const size_t zbase = (size_t)b * CDIM * SPAT + s0;

    // ---- stage z tile: [point][channel] bf16, coalesced global reads ----
    for (int e = tid; e < PTS_PER_BLOCK * CDIM; e += 128) {
        int c = e >> 7;          // /128
        int p = e & 127;
        z_lds[p * ZSTRIDE + c] = (__bf16)z[zbase + (size_t)c * SPAT + p];
    }
    __syncthreads();

    const int lane    = tid & 31;
    const int wave    = tid >> 5;
    const int lrow    = lane & 15;
    const int upper   = (lane >= 16) ? 1 : 0;
    const int coffA   = upper * 16;               // A half-lane K offset
    const int coffB   = upper * 16;               // B half-lane K offset
    const __bf16* zrow0 = &z_lds[(wave * 32 + lrow) * ZSTRIDE];
    const __bf16* zrow1 = &z_lds[(wave * 32 + 16 + lrow) * ZSTRIDE];

    float minv[16];
    int   mini[16];
#pragma unroll
    for (int r = 0; r < 16; ++r) { minv[r] = 3.4e38f; mini[r] = 0; }

    for (int n0 = 0; n0 < KCODES; n0 += NCHUNK) {
        // ---- cooperatively stage NCHUNK codebook rows (bf16) into LDS ----
        const unsigned int* esrc = (const unsigned int*)(emb_bf + (size_t)n0 * CDIM);
        unsigned int* edst = (unsigned int*)e_lds;
        for (int e = tid; e < NCHUNK * (CDIM / 2); e += 128) {
            int code = e >> 7;        // /128 dwords per row
            int c2   = e & 127;
            edst[code * (ZSTRIDE / 2) + c2] = esrc[code * (CDIM / 2) + c2];
        }
        __syncthreads();

        v8f acc[8];
#pragma unroll
        for (int q = 0; q < 8; ++q) acc[q] = (v8f){};

#pragma unroll
        for (int k0 = 0; k0 < CDIM; k0 += 32) {
            // A tiles (16x32 bf16): elements 0-15 = K[coffA..+15]
            v8bf a0lo = *(const v8bf*)(zrow0 + k0 + coffA);
            v8bf a0hi = *(const v8bf*)(zrow0 + k0 + coffA + 8);
            v16bf a0 = __builtin_shufflevector(a0lo, a0hi,
                0,1,2,3,4,5,6,7,8,9,10,11,12,13,14,15);
            v8bf a1lo = *(const v8bf*)(zrow1 + k0 + coffA);
            v8bf a1hi = *(const v8bf*)(zrow1 + k0 + coffA + 8);
            v16bf a1 = __builtin_shufflevector(a1lo, a1hi,
                0,1,2,3,4,5,6,7,8,9,10,11,12,13,14,15);

            // 4 B tiles (32x16 bf16): 16 contiguous K per lane half
#pragma unroll
            for (int t = 0; t < 4; ++t) {
                const __bf16* erow = &e_lds[(t * 16 + lrow) * ZSTRIDE];
                v8bf blo = *(const v8bf*)(erow + k0 + coffB);
                v8bf bhi = *(const v8bf*)(erow + k0 + coffB + 8);
                v16bf bt = __builtin_shufflevector(blo, bhi,
                    0,1,2,3,4,5,6,7,8,9,10,11,12,13,14,15);
                acc[2 * t] = __builtin_amdgcn_wmma_f32_16x16x32_bf16(
                    false, a0, false, bt, (short)0, acc[2 * t], false, false);
                acc[2 * t + 1] = __builtin_amdgcn_wmma_f32_16x16x32_bf16(
                    false, a1, false, bt, (short)0, acc[2 * t + 1], false, false);
            }
        }

        // ---- d = ||e||^2 - 2*dot ; running argmin (||z||^2 constant per row) ----
#pragma unroll
        for (int t = 0; t < 4; ++t) {
            int n = n0 + t * 16 + lrow;
            float ne = norms_e[n];
#pragma unroll
            for (int r = 0; r < 8; ++r) {
                float d0 = ne - 2.0f * acc[2 * t][r];
                if (d0 < minv[r]) { minv[r] = d0; mini[r] = n; }
                float d1 = ne - 2.0f * acc[2 * t + 1][r];
                if (d1 < minv[8 + r]) { minv[8 + r] = d1; mini[8 + r] = n; }
            }
        }
        __syncthreads();   // e_lds reused next iteration
    }

    // ---- cross-lane argmin within each 16-lane half ----
#pragma unroll
    for (int off = 1; off < 16; off <<= 1) {
#pragma unroll
        for (int r = 0; r < 16; ++r) {
            float ov = __shfl_xor(minv[r], off, 32);
            int   oi = __shfl_xor(mini[r], off, 32);
            if (ov < minv[r]) { minv[r] = ov; mini[r] = oi; }
        }
    }
    if (lrow == 0) {
#pragma unroll
        for (int m = 0; m < 2; ++m) {
            int rowbase = wave * 32 + m * 16 + upper * 8;
#pragma unroll
            for (int r = 0; r < 8; ++r) {
                int pt  = rowbase + r;
                int ci  = mini[m * 8 + r];
                idx_lds[pt] = ci;
                idx_out[g0 + pt] = (float)ci;
                atomicAdd(&counts[ci], 1u);
            }
        }
    }
    __syncthreads();

    // ---- gather zq (exact fp32 codebook), write output, accumulate loss ----
    float lsum = 0.f;
    for (int e = tid; e < PTS_PER_BLOCK * CDIM; e += 128) {
        int c = e >> 7;
        int p = e & 127;
        size_t gi = zbase + (size_t)c * SPAT + p;
        float val = emb[(size_t)idx_lds[p] * CDIM + c];
        float df  = val - z[gi];
        lsum += df * df;
        zq_out[gi] = val;   // straight-through forward value == zq
    }
    red_lds[tid] = lsum;
    __syncthreads();
    for (int s = 64; s > 0; s >>= 1) {
        if (tid < s) red_lds[tid] += red_lds[tid + s];
        __syncthreads();
    }
    if (tid == 0) atomicAdd(loss_acc, red_lds[0]);
}

// ============================================================
// Kernel 5: scalars — loss, perplexity, mean_distance
// ============================================================
__global__ __launch_bounds__(256) void finalize(const float* __restrict__ sz,
                                                const float* __restrict__ se,
                                                const float* __restrict__ Sz,
                                                const float* __restrict__ Se,
                                                const float* __restrict__ loss_acc,
                                                const unsigned int* __restrict__ counts,
                                                float* __restrict__ out_loss,
                                                float* __restrict__ out_perp,
                                                float* __restrict__ out_md) {
    __shared__ float red[256];
    const int t = threadIdx.x;

    // cross = sum_c sz[c]*se[c]
    red[t] = sz[t] * se[t];
    __syncthreads();
    for (int s = 128; s > 0; s >>= 1) {
        if (t < s) red[t] += red[t + s];
        __syncthreads();
    }
    float cross = red[0];
    __syncthreads();

    // entropy of code usage
    float h = 0.f;
    for (int k = t; k < KCODES; k += 256) {
        float em = (float)counts[k] * (1.0f / (float)NPTS);
        h += em * logf(em + 1e-10f);
    }
    red[t] = h;
    __syncthreads();
    for (int s = 128; s > 0; s >>= 1) {
        if (t < s) red[t] += red[t + s];
        __syncthreads();
    }
    if (t == 0) {
        out_perp[0] = expf(-red[0]);
        out_loss[0] = 1.25f * loss_acc[0] * (1.0f / (float)TOTELM);
        out_md[0] = ((float)KCODES * Sz[0] + (float)NPTS * Se[0] - 2.0f * cross)
                    * (1.0f / ((float)NPTS * (float)KCODES));
    }
}

// ============================================================
extern "C" void kernel_launch(void* const* d_in, const int* in_sizes, int n_in,
                              void* d_out, int out_size, void* d_ws, size_t ws_size,
                              hipStream_t stream) {
    (void)in_sizes; (void)n_in; (void)out_size; (void)ws_size;
    const float* z   = (const float*)d_in[0];   // [8,256,16,16,16]
    const float* emb = (const float*)d_in[1];   // [1024,256]

    char* ws = (char*)d_ws;
    __bf16*       emb_bf   = (__bf16*)(ws + WS_EMB_BF);
    float*        norms_e  = (float*)(ws + WS_NORMS_E);
    float*        se       = (float*)(ws + WS_SE);
    float*        Se       = (float*)(ws + WS_SE_TOT);
    unsigned int* counts   = (unsigned int*)(ws + WS_COUNTS);
    float*        sz       = (float*)(ws + WS_SZ);
    float*        Sz       = (float*)(ws + WS_SZ_TOT);
    float*        loss_acc = (float*)(ws + WS_LOSS);

    float* out      = (float*)d_out;
    float* zq_out   = out;                      // [8,256,16,16,16]
    float* out_loss = out + TOTELM;             // scalar
    float* out_perp = out + TOTELM + 1;         // scalar
    float* idx_out  = out + TOTELM + 2;         // [32768] as float
    float* out_md   = out + TOTELM + 2 + NPTS;  // scalar

    hipMemsetAsync(ws + WS_ZERO0, 0, WS_ZERO_LEN, stream);

    prep_emb<<<KCODES, 256, 0, stream>>>(emb, emb_bf, norms_e);
    emb_stats<<<1, 256, 0, stream>>>(emb, se, Se);
    z_stats<<<8 * CDIM, 256, 0, stream>>>(z, sz, Sz);
    vq_main<<<NPTS / PTS_PER_BLOCK, 128, 0, stream>>>(
        z, emb, emb_bf, norms_e, zq_out, idx_out, counts, loss_acc);
    finalize<<<1, 256, 0, stream>>>(sz, se, Sz, Se, loss_acc, counts,
                                    out_loss, out_perp, out_md);
}
